// TrackEmbedder_36129264894667
// MI455X (gfx1250) — matrix operand order repaired and checked by reading, not compiled, not run on hardware
//
#include <hip/hip_runtime.h>
#include <hip/hip_bf16.h>
#include <math.h>

typedef __attribute__((ext_vector_type(16))) _Float16 v16h;
typedef __attribute__((ext_vector_type(8)))  float    v8f;

#define NTRK_TOT 96000
#define LHD 128          // lstm hidden
#define G4  512          // 4*LHD gates
#define HIDD 256         // mlp hidden/out

// Fast activations: single v_rcp_f32 + v_exp_f32 instead of the
// correctly-rounded division / libm-tanh sequences.
__device__ __forceinline__ float fast_rcp(float x) { return __builtin_amdgcn_rcpf(x); }
__device__ __forceinline__ float sigmf(float x) { return fast_rcp(1.0f + __expf(-x)); }
__device__ __forceinline__ float tanh_fast(float x) {
  // tanh(x) = 1 - 2/(exp(2x)+1); saturates correctly: exp->inf => 1, exp->0 => -1
  float e = __expf(2.0f * x);
  return 1.0f - 2.0f * fast_rcp(e + 1.0f);
}

// A fragment (16x32 f16, M x K), row-major source with leading dim ldk.
// ISA layout: lanes 0-15 hold row M=lane, K = kb+{0..7,16..23}; lanes 16-31 K = kb+{8..15,24..31}.
__device__ __forceinline__ v16h load_a16(const _Float16* base, int ldk, int kb, int lane) {
  int m    = lane & 15;
  int koff = kb + ((lane < 16) ? 0 : 8);
  const _Float16* p = base + m * ldk + koff;
  v16h a;
#pragma unroll
  for (int i = 0; i < 8; ++i) { a[i] = p[i]; a[i + 8] = p[i + 16]; }
  return a;
}

// B fragment (32x16 f16, K x N) where B = W^T and W is [N][K] row-major:
// lane's column N = n0 + (lane&15) is read from W's row.
// lanes 0-15 hold K = kb+{0..7,16..23}, lanes 16-31 hold K = kb+{8..15,24..31}.
__device__ __forceinline__ v16h load_b16(const _Float16* W, int ldk, int n0, int kb, int lane) {
  int n = n0 + (lane & 15);
  int k = kb + ((lane < 16) ? 0 : 8);
  const _Float16* p = W + n * ldk + k;
  v16h b;
#pragma unroll
  for (int i = 0; i < 8; ++i) { b[i] = p[i]; b[i + 8] = p[i + 16]; }
  return b;
}

__device__ __forceinline__ v8f wmma_f16(v16h a, v16h b, v8f c) {
  return __builtin_amdgcn_wmma_f32_16x16x32_f16(false, a, false, b, (short)0, c, false, false);
}

// ---------------------------------------------------------------------------
// Kernel 1: f32 -> f16 weight conversion + bias fusion into workspace
// ---------------------------------------------------------------------------
__global__ __launch_bounds__(256) void convert_weights(
    const float* __restrict__ Whh0, const float* __restrict__ Wih1,
    const float* __restrict__ Whh1, const float* __restrict__ w0,
    const float* __restrict__ w1,
    const float* __restrict__ bih0, const float* __restrict__ bhh0,
    const float* __restrict__ bih1, const float* __restrict__ bhh1,
    _Float16* __restrict__ Whh0h, _Float16* __restrict__ Wih1h,
    _Float16* __restrict__ Whh1h, _Float16* __restrict__ W0h,
    _Float16* __restrict__ W1h, float* __restrict__ bias0,
    float* __restrict__ bias1)
{
  int idx = blockIdx.x * blockDim.x + threadIdx.x;
  if (idx < 65536) { Whh0h[idx] = (_Float16)Whh0[idx]; return; }
  idx -= 65536;
  if (idx < 65536) { Wih1h[idx] = (_Float16)Wih1[idx]; return; }
  idx -= 65536;
  if (idx < 65536) { Whh1h[idx] = (_Float16)Whh1[idx]; return; }
  idx -= 65536;
  if (idx < 32768) {             // mlp_w0 columns 3..130 -> [256][128]
    int n = idx >> 7, k = idx & 127;
    W0h[idx] = (_Float16)w0[n * 131 + 3 + k];
    return;
  }
  idx -= 32768;
  if (idx < 65536) { W1h[idx] = (_Float16)w1[idx]; return; }
  idx -= 65536;
  if (idx < 512) { bias0[idx] = bih0[idx] + bhh0[idx]; return; }
  idx -= 512;
  if (idx < 512) { bias1[idx] = bih1[idx] + bhh1[idx]; return; }
}

// ---------------------------------------------------------------------------
// Kernel 2: fused 2-layer LSTM + 2-layer MLP, 16 sequences per block.
// 8 wave32s; each wave owns 4 gate N-tiles (512 gates / 16 / 8).
// ---------------------------------------------------------------------------
__global__ __launch_bounds__(256) void track_lstm_mlp(
    const float* __restrict__ in0,     // [64][18][1500]
    const float* __restrict__ Wih0,    // [512][2]  f32 (K=2 folded on VALU)
    const float* __restrict__ mlp_w0,  // [256][131] f32 (cols 0..2 on VALU)
    const float* __restrict__ mlp_b0,  // [256]
    const float* __restrict__ mlp_b1,  // [256]
    const _Float16* __restrict__ Whh0h,
    const _Float16* __restrict__ Wih1h,
    const _Float16* __restrict__ Whh1h,
    const _Float16* __restrict__ W0h,  // [256][128]
    const _Float16* __restrict__ W1h,  // [256][256]
    const float* __restrict__ bias0,   // [512]
    const float* __restrict__ bias1,   // [512]
    float* __restrict__ fe_out)        // [96000][256]
{
  __shared__ _Float16 h_lds[16][LHD];       // current hidden, f16 (A operand)
  __shared__ float    c_lds[16][LHD];       // cell state
  __shared__ float    gates[16][G4];        // gate pre-activations
  __shared__ _Float16 hs0[6][16][LHD];      // layer-0 hidden states, all steps
  __shared__ float    xfeat[16][16];        // [0..2]=tf0..2, [3..14]=traj
  __shared__ _Float16 hml[16][HIDD];        // relu(mlp hidden), f16

  const int tid   = threadIdx.x;
  const int lane  = tid & 31;
  const int wave  = tid >> 5;
  const int sbase = blockIdx.x * 16;
  const int mb    = (lane < 16) ? 0 : 8;    // C/D row base per lane half
  const int nl    = lane & 15;              // C/D col within tile

  // gather per-track features (coalesced across the 16 rows of the block)
  for (int t = tid; t < 16 * 15; t += 256) {
    int row = t / 15, f = t % 15;
    int s = sbase + row;
    int b = s / 1500, tr = s % 1500;
    int feat = (f < 3) ? f : (f + 3);       // f>=3 -> features 6..17 (traj)
    xfeat[row][f] = in0[b * 27000 + feat * 1500 + tr];
  }
  for (int e = tid; e < 16 * LHD; e += 256) {
    int m = e >> 7, j = e & 127;
    h_lds[m][j] = (_Float16)0.0f;
    c_lds[m][j] = 0.0f;
  }
  __syncthreads();

  // ---------------- LSTM layer 0 (input size 2) ----------------
  for (int t = 0; t < 6; ++t) {
#pragma unroll
    for (int tile = 0; tile < 4; ++tile) {
      const int n0 = (wave * 4 + tile) * 16;
      const int n  = n0 + nl;
      const float wa = Wih0[n * 2 + 0];
      const float wb = Wih0[n * 2 + 1];
      const float bb = bias0[n];
      v8f acc;
#pragma unroll
      for (int v = 0; v < 8; ++v) {
        int m = mb + v;
        acc[v] = bb + xfeat[m][3 + 2 * t] * wa + xfeat[m][4 + 2 * t] * wb;
      }
#pragma unroll
      for (int kb = 0; kb < LHD; kb += 32) {
        v16h a = load_a16(&h_lds[0][0], LHD, kb, lane);
        v16h b = load_b16(Whh0h, LHD, n0, kb, lane);
        acc = wmma_f16(a, b, acc);
      }
#pragma unroll
      for (int v = 0; v < 8; ++v) gates[mb + v][n] = acc[v];
    }
    __syncthreads();
    for (int e = tid; e < 16 * LHD; e += 256) {
      int m = e >> 7, j = e & 127;
      float ig = gates[m][j];
      float fg = gates[m][128 + j];
      float gg = gates[m][256 + j];
      float og = gates[m][384 + j];
      float c  = sigmf(fg) * c_lds[m][j] + sigmf(ig) * tanh_fast(gg);
      float h  = sigmf(og) * tanh_fast(c);
      c_lds[m][j]   = c;
      h_lds[m][j]   = (_Float16)h;
      hs0[t][m][j]  = (_Float16)h;
    }
    __syncthreads();
  }

  // reset state for layer 1
  for (int e = tid; e < 16 * LHD; e += 256) {
    int m = e >> 7, j = e & 127;
    h_lds[m][j] = (_Float16)0.0f;
    c_lds[m][j] = 0.0f;
  }
  __syncthreads();

  // ---------------- LSTM layer 1 (input size 128) ----------------
  for (int t = 0; t < 6; ++t) {
#pragma unroll
    for (int tile = 0; tile < 4; ++tile) {
      const int n0 = (wave * 4 + tile) * 16;
      const int n  = n0 + nl;
      const float bb = bias1[n];
      v8f acc;
#pragma unroll
      for (int v = 0; v < 8; ++v) acc[v] = bb;
#pragma unroll
      for (int kb = 0; kb < LHD; kb += 32) {     // x-part: hs0[t] @ Wih1^T
        v16h a = load_a16(&hs0[t][0][0], LHD, kb, lane);
        v16h b = load_b16(Wih1h, LHD, n0, kb, lane);
        acc = wmma_f16(a, b, acc);
      }
#pragma unroll
      for (int kb = 0; kb < LHD; kb += 32) {     // h-part: h1 @ Whh1^T
        v16h a = load_a16(&h_lds[0][0], LHD, kb, lane);
        v16h b = load_b16(Whh1h, LHD, n0, kb, lane);
        acc = wmma_f16(a, b, acc);
      }
#pragma unroll
      for (int v = 0; v < 8; ++v) gates[mb + v][n] = acc[v];
    }
    __syncthreads();
    for (int e = tid; e < 16 * LHD; e += 256) {
      int m = e >> 7, j = e & 127;
      float ig = gates[m][j];
      float fg = gates[m][128 + j];
      float gg = gates[m][256 + j];
      float og = gates[m][384 + j];
      float c  = sigmf(fg) * c_lds[m][j] + sigmf(ig) * tanh_fast(gg);
      float h  = sigmf(og) * tanh_fast(c);
      c_lds[m][j] = c;
      h_lds[m][j] = (_Float16)h;
    }
    __syncthreads();
  }

  // ---------------- MLP layer 0: relu([tf0..2, h1] @ w0^T + b0) ----------------
#pragma unroll
  for (int tile = 0; tile < 2; ++tile) {
    const int n0 = (wave * 2 + tile) * 16;
    const int n  = n0 + nl;
    const float w0a = mlp_w0[n * 131 + 0];
    const float w0b = mlp_w0[n * 131 + 1];
    const float w0c = mlp_w0[n * 131 + 2];
    const float bb  = mlp_b0[n];
    v8f acc;
#pragma unroll
    for (int v = 0; v < 8; ++v) {
      int m = mb + v;
      acc[v] = bb + xfeat[m][0] * w0a + xfeat[m][1] * w0b + xfeat[m][2] * w0c;
    }
#pragma unroll
    for (int kb = 0; kb < LHD; kb += 32) {
      v16h a = load_a16(&h_lds[0][0], LHD, kb, lane);
      v16h b = load_b16(W0h, LHD, n0, kb, lane);
      acc = wmma_f16(a, b, acc);
    }
#pragma unroll
    for (int v = 0; v < 8; ++v) {
      float r = acc[v];
      hml[mb + v][n] = (_Float16)(r > 0.0f ? r : 0.0f);
    }
  }
  __syncthreads();

  // ---------------- MLP layer 1: h @ w1^T + b1 -> final_embedding ----------------
#pragma unroll
  for (int tile = 0; tile < 2; ++tile) {
    const int n0 = (wave * 2 + tile) * 16;
    const int n  = n0 + nl;
    const float bb = mlp_b1[n];
    v8f acc;
#pragma unroll
    for (int v = 0; v < 8; ++v) acc[v] = bb;
#pragma unroll
    for (int kb = 0; kb < HIDD; kb += 32) {
      v16h a = load_a16(&hml[0][0], HIDD, kb, lane);
      v16h b = load_b16(W1h, HIDD, n0, kb, lane);
      acc = wmma_f16(a, b, acc);
    }
#pragma unroll
    for (int v = 0; v < 8; ++v)
      fe_out[(size_t)(sbase + mb + v) * 256 + n] = acc[v];
  }
}

// ---------------------------------------------------------------------------
// Kernel 3: DETR sine/cos positional embedding + eta/phi/logpt passthrough
// ---------------------------------------------------------------------------
__global__ __launch_bounds__(256) void posenc_kernel(
    const float* __restrict__ in0, float* __restrict__ pos,
    float* __restrict__ logpt_o, float* __restrict__ eta_o,
    float* __restrict__ phi_o)
{
  int idx = blockIdx.x * blockDim.x + threadIdx.x;    // 96000*64 threads
  if (idx >= NTRK_TOT * 64) return;
  int s = idx >> 6;
  int j = idx & 63;
  int b = s / 1500, tr = s % 1500;
  float eta = in0[b * 27000 + 3 * 1500 + tr];
  float phi = in0[b * 27000 + 4 * 1500 + tr];
  float x = (j < 32) ? eta : phi;
  int k = j & 31;
  // dim_t = 10000^(k/32); invt = exp(-k*ln(10000)/32) -> one v_exp_f32
  float invt = __expf(-(float)k * (9.210340371976184f / 32.0f));
  float p = x * 6.283185307179586f * invt;
  int base = s * 128 + ((j < 32) ? 0 : 64) + 2 * k;
  pos[base]     = __sinf(p);
  pos[base + 1] = __cosf(p);
  if (j == 0) {
    logpt_o[s] = in0[b * 27000 + 2 * 1500 + tr];
    eta_o[s]   = eta;
    phi_o[s]   = phi;
  }
}

// ---------------------------------------------------------------------------
extern "C" void kernel_launch(void* const* d_in, const int* in_sizes, int n_in,
                              void* d_out, int out_size, void* d_ws, size_t ws_size,
                              hipStream_t stream) {
  const float* in0  = (const float*)d_in[0];   // track_feats [64][18][1500]
  const float* Wih0 = (const float*)d_in[1];   // [512][2]
  const float* Whh0 = (const float*)d_in[2];   // [512][128]
  const float* bih0 = (const float*)d_in[3];
  const float* bhh0 = (const float*)d_in[4];
  const float* Wih1 = (const float*)d_in[5];   // [512][128]
  const float* Whh1 = (const float*)d_in[6];   // [512][128]
  const float* bih1 = (const float*)d_in[7];
  const float* bhh1 = (const float*)d_in[8];
  const float* w0   = (const float*)d_in[9];   // [256][131]
  const float* b0   = (const float*)d_in[10];
  const float* w1   = (const float*)d_in[11];  // [256][256]
  const float* b1   = (const float*)d_in[12];

  // workspace layout (f16 weights + fused biases), ~580 KB
  _Float16* Whh0h = (_Float16*)d_ws;
  _Float16* Wih1h = Whh0h + 512 * 128;
  _Float16* Whh1h = Wih1h + 512 * 128;
  _Float16* W0h   = Whh1h + 512 * 128;
  _Float16* W1h   = W0h   + 256 * 128;
  float*    bias0 = (float*)(W1h + 256 * 256);
  float*    bias1 = bias0 + 512;

  // 1) weight conversion: 295936 work items, exact multiple of 256
  convert_weights<<<295936 / 256, 256, 0, stream>>>(
      Whh0, Wih1, Whh1, w0, w1, bih0, bhh0, bih1, bhh1,
      Whh0h, Wih1h, Whh1h, W0h, W1h, bias0, bias1);

  // output layout: FE | POS | logpt | eta | phi
  float* fe    = (float*)d_out;
  float* pos   = fe  + (size_t)NTRK_TOT * 256;
  float* logpt = pos + (size_t)NTRK_TOT * 128;
  float* eta   = logpt + NTRK_TOT;
  float* phi   = eta   + NTRK_TOT;

  // 2) fused LSTM+MLP: 96000 tracks / 16 per block = 6000 blocks
  track_lstm_mlp<<<6000, 256, 0, stream>>>(
      in0, Wih0, w0, b0, b1, Whh0h, Wih1h, Whh1h, W0h, W1h, bias0, bias1, fe);

  // 3) positional encodings + scalar passthroughs
  posenc_kernel<<<(NTRK_TOT * 64) / 256, 256, 0, stream>>>(in0, pos, logpt, eta, phi);
}
